// SelfAlignment_32238024524113
// MI455X (gfx1250) — matrix operand order repaired and checked by reading, not compiled, not run on hardware
//
#include <hip/hip_runtime.h>

typedef unsigned short u16;
typedef unsigned int   u32;

typedef __attribute__((ext_vector_type(16))) __bf16 v16bf;
typedef __attribute__((ext_vector_type(8)))  float  v8f;

#define BB 8
#define SS 2048
#define DD 1024
#define LDSTR 40   // LDS row stride in bf16 elements (padded, conflict-free, 16B aligned)

// ---- bf16 conversion: hardware v_cvt_pk_bf16_f32 (mnemonic confirmed on gfx1250)

__device__ __forceinline__ u32 pk(float a, float b) {
    u32 r;
    asm("v_cvt_pk_bf16_f32 %0, %1, %2" : "=v"(r) : "v"(a), "v"(b));
    return r;   // lo = bf16(a), hi = bf16(b)
}
__device__ __forceinline__ u16 f2bf(float f) {
    return (u16)(pk(f, f) & 0xFFFFu);
}

// ---- WMMA fragments --------------------------------------------------------

union FragU { uint4 q[2]; v16bf v; };

// A fragment, 16x32 bf16 (ISA layout):
// lanes 0-15: M=lane,    K chunks [0..7]  and [16..23]
// lanes16-31: M=lane-16, K chunks [8..15] and [24..31]
__device__ __forceinline__ v16bf load_frag_a(const u16* lds, int row0) {
    int lane = threadIdx.x & 31;
    int m  = row0 + (lane & 15);
    int k0 = (lane < 16) ? 0 : 8;
    const u16* p = lds + m * LDSTR + k0;
    FragU u;
    u.q[0] = *(const uint4*)(p);
    u.q[1] = *(const uint4*)(p + 16);
    return u.v;
}

// B fragment, 32x16 bf16: lane n = lane&15, K chunks [k0..k0+7] and [k0+16..k0+23], k0 = (lane<16 ? 0 : 8)
__device__ __forceinline__ v16bf load_frag_b(const u16* lds, int col0) {
    int lane = threadIdx.x & 31;
    int n  = col0 + (lane & 15);
    int k0 = (lane < 16) ? 0 : 8;
    const u16* p = lds + n * LDSTR + k0;
    FragU u;
    u.q[0] = *(const uint4*)(p);
    u.q[1] = *(const uint4*)(p + 16);
    return u.v;
}

// One K=32 step: each wave computes a 32x64 tile = 2x4 WMMAs.
__device__ __forceinline__ void mma_tile(const u16* lA, const u16* lB, v8f acc[2][4]) {
    int wid = threadIdx.x >> 5;
    int wm = (wid >> 1) * 32;
    int wn = (wid & 1) * 64;
    v16bf a[2], b[4];
#pragma unroll
    for (int i = 0; i < 2; i++) a[i] = load_frag_a(lA, wm + i * 16);
#pragma unroll
    for (int j = 0; j < 4; j++) b[j] = load_frag_b(lB, wn + j * 16);
#pragma unroll
    for (int i = 0; i < 2; i++)
#pragma unroll
        for (int j = 0; j < 4; j++)
            acc[i][j] = __builtin_amdgcn_wmma_f32_16x16x32_bf16(
                false, a[i], false, b[j], (short)0, acc[i][j], false, false);
}

// ---- global fetch / LDS store (pipelined halves) ---------------------------

struct TileF32 { float4 x[4]; };   // 128x32 fp32 row-major tile, 16 floats/thread
struct TileB16 { uint4 lo, hi; };  // 128x32 bf16 row-major tile, 16 bf16/thread

__device__ __forceinline__ TileF32 fetch_f32(const float* __restrict__ g, int ld,
                                             int rowBase, int kBase) {
    int t = threadIdx.x;
    int r = t >> 1;
    int c = (t & 1) << 4;
    const float* src = g + (size_t)(rowBase + r) * ld + kBase + c;
    TileF32 T;
    T.x[0] = ((const float4*)src)[0];
    T.x[1] = ((const float4*)src)[1];
    T.x[2] = ((const float4*)src)[2];
    T.x[3] = ((const float4*)src)[3];
    __builtin_prefetch(src + 32, 0, 1);   // hint: next K tile
    return T;
}
__device__ __forceinline__ void store_f32(const TileF32& T, u16* lds) {
    int t = threadIdx.x;
    int r = t >> 1;
    int c = (t & 1) << 4;
    uint4 lo, hi;
    lo.x = pk(T.x[0].x, T.x[0].y); lo.y = pk(T.x[0].z, T.x[0].w);
    lo.z = pk(T.x[1].x, T.x[1].y); lo.w = pk(T.x[1].z, T.x[1].w);
    hi.x = pk(T.x[2].x, T.x[2].y); hi.y = pk(T.x[2].z, T.x[2].w);
    hi.z = pk(T.x[3].x, T.x[3].y); hi.w = pk(T.x[3].z, T.x[3].w);
    uint4* dst = (uint4*)(lds + r * LDSTR + c);
    dst[0] = lo; dst[1] = hi;
}

__device__ __forceinline__ TileB16 fetch_bf16(const u16* __restrict__ g, int ld,
                                              int rowBase, int kBase) {
    int t = threadIdx.x;
    int r = t >> 1;
    int c = (t & 1) << 4;
    const u16* src = g + (size_t)(rowBase + r) * ld + kBase + c;
    TileB16 T;
    T.lo = ((const uint4*)src)[0];
    T.hi = ((const uint4*)src)[1];
    __builtin_prefetch(src + 32, 0, 1);
    return T;
}
__device__ __forceinline__ void store_bf16(const TileB16& T, u16* lds) {
    int t = threadIdx.x;
    int r = t >> 1;
    int c = (t & 1) << 4;
    uint4* dst = (uint4*)(lds + r * LDSTR + c);
    dst[0] = T.lo; dst[1] = T.hi;
}

// ---- kernels ---------------------------------------------------------------

// K0: one-pass prep: text1 -> bf16 row-major AND bf16 transposed (via LDS tile)
__global__ void __launch_bounds__(256)
k_prep_text1(const float* __restrict__ text1, u16* __restrict__ t1bf,
             u16* __restrict__ t1bfT) {
    __shared__ u16 tile[64][65];
    int b  = blockIdx.z;
    int s0 = blockIdx.y * 64;
    int d0 = blockIdx.x * 64;
    const float* src = text1 + (size_t)b * SS * DD;
    u16* dst  = t1bf  + (size_t)b * SS * DD;
    u16* dstT = t1bfT + (size_t)b * DD * SS;
    int t  = threadIdx.x;
    int r  = t >> 4;          // 0..15
    int c4 = (t & 15) << 2;   // 0,4,..,60
#pragma unroll
    for (int i = 0; i < 4; i++) {
        int row = r + 16 * i;
        float4 x = *(const float4*)(src + (size_t)(s0 + row) * DD + d0 + c4);
        u32 p0 = pk(x.x, x.y), p1 = pk(x.z, x.w);
        *(uint2*)(dst + (size_t)(s0 + row) * DD + d0 + c4) = make_uint2(p0, p1);
        tile[row][c4 + 0] = (u16)(p0 & 0xFFFFu);
        tile[row][c4 + 1] = (u16)(p0 >> 16);
        tile[row][c4 + 2] = (u16)(p1 & 0xFFFFu);
        tile[row][c4 + 3] = (u16)(p1 >> 16);
    }
    __syncthreads();
#pragma unroll
    for (int i = 0; i < 4; i++) {
        int drow = r + 16 * i;   // row in transposed = col in original
        u32 p0 = (u32)tile[c4 + 0][drow] | ((u32)tile[c4 + 1][drow] << 16);
        u32 p1 = (u32)tile[c4 + 2][drow] | ((u32)tile[c4 + 3][drow] << 16);
        *(uint2*)(dstT + (size_t)(d0 + drow) * SS + s0 + c4) = make_uint2(p0, p1);
    }
}

// K1: q_bf16 = bf16(text @ W^T)    M=B*S, N=D, K=D
__global__ void __launch_bounds__(256)
k_gemm_q(const float* __restrict__ text, const float* __restrict__ W,
         u16* __restrict__ qbf) {
    __shared__ __align__(16) u16 lA[2][128 * LDSTR];
    __shared__ __align__(16) u16 lB[2][128 * LDSTR];
    int mBase = blockIdx.y * 128;
    int nBase = blockIdx.x * 128;
    v8f acc[2][4] = {};

    TileF32 ta = fetch_f32(text, DD, mBase, 0);
    TileF32 tb = fetch_f32(W,    DD, nBase, 0);
    store_f32(ta, lA[0]); store_f32(tb, lB[0]);
    __syncthreads();
    int cur = 0;
    for (int kk = 0; kk < DD; kk += 32) {
        bool more = (kk + 32) < DD;
        TileF32 na, nb;
        if (more) {
            na = fetch_f32(text, DD, mBase, kk + 32);
            nb = fetch_f32(W,    DD, nBase, kk + 32);
        }
        mma_tile(lA[cur], lB[cur], acc);
        if (more) {
            store_f32(na, lA[cur ^ 1]);
            store_f32(nb, lB[cur ^ 1]);
        }
        __syncthreads();
        cur ^= 1;
    }

    int lane = threadIdx.x & 31;
    int wid  = threadIdx.x >> 5;
    int wm = (wid >> 1) * 32, wn = (wid & 1) * 64;
#pragma unroll
    for (int i = 0; i < 2; i++)
#pragma unroll
        for (int j = 0; j < 4; j++) {
            int col  = nBase + wn + j * 16 + (lane & 15);
            int row0 = mBase + wm + i * 16 + ((lane >> 4) & 1) * 8;
#pragma unroll
            for (int e = 0; e < 8; e++)
                qbf[(size_t)(row0 + e) * DD + col] = f2bf(acc[i][j][e]);
        }
}

// K2: logits[b] = q[b] @ text1[b]^T  (raw fp32, into d_out attn region)
__global__ void __launch_bounds__(256)
k_gemm_logits(const u16* __restrict__ qbf, const u16* __restrict__ t1bf,
              float* __restrict__ logits) {
    __shared__ __align__(16) u16 lA[2][128 * LDSTR];
    __shared__ __align__(16) u16 lB[2][128 * LDSTR];
    int b = blockIdx.z;
    const u16* A  = qbf    + (size_t)b * SS * DD;
    const u16* Bt = t1bf   + (size_t)b * SS * DD;
    float*     C  = logits + (size_t)b * SS * SS;
    int mBase = blockIdx.y * 128;
    int nBase = blockIdx.x * 128;
    v8f acc[2][4] = {};

    TileB16 ta = fetch_bf16(A,  DD, mBase, 0);
    TileB16 tb = fetch_bf16(Bt, DD, nBase, 0);
    store_bf16(ta, lA[0]); store_bf16(tb, lB[0]);
    __syncthreads();
    int cur = 0;
    for (int kk = 0; kk < DD; kk += 32) {
        bool more = (kk + 32) < DD;
        TileB16 na, nb;
        if (more) {
            na = fetch_bf16(A,  DD, mBase, kk + 32);
            nb = fetch_bf16(Bt, DD, nBase, kk + 32);
        }
        mma_tile(lA[cur], lB[cur], acc);
        if (more) {
            store_bf16(na, lA[cur ^ 1]);
            store_bf16(nb, lB[cur ^ 1]);
        }
        __syncthreads();
        cur ^= 1;
    }

    int lane = threadIdx.x & 31;
    int wid  = threadIdx.x >> 5;
    int wm = (wid >> 1) * 32, wn = (wid & 1) * 64;
#pragma unroll
    for (int i = 0; i < 2; i++)
#pragma unroll
        for (int j = 0; j < 4; j++) {
            int col  = nBase + wn + j * 16 + (lane & 15);
            int row0 = mBase + wm + i * 16 + ((lane >> 4) & 1) * 8;
#pragma unroll
            for (int e = 0; e < 8; e++)
                C[(size_t)(row0 + e) * SS + col] = acc[i][j][e];
        }
}

// K3: masked softmax per row, in place, scaled by query mask.
__global__ void __launch_bounds__(256)
k_softmax(float* __restrict__ attn, const float* __restrict__ mask) {
    int rowId = blockIdx.x;              // 0 .. B*S-1
    int b    = rowId >> 11;
    int srow = rowId & (SS - 1);
    float* rp = attn + (size_t)rowId * SS;
    const float* mk = mask + (size_t)b * SS;
    float maskq = mk[srow];
    __shared__ float red[256];
    int t = threadIdx.x;
    float vals[8];
    float mloc = -3.0e38f;
#pragma unroll
    for (int e = 0; e < 8; e++) {
        int idx = t + e * 256;
        float l  = rp[idx];
        float km = mk[idx];
        float v  = (km > 0.0f) ? l : -3.0e38f;
        vals[e] = v;
        mloc = fmaxf(mloc, v);
    }
    red[t] = mloc; __syncthreads();
    for (int s = 128; s > 0; s >>= 1) {
        if (t < s) red[t] = fmaxf(red[t], red[t + s]);
        __syncthreads();
    }
    float rmax = red[0]; __syncthreads();
    float sloc = 0.0f;
#pragma unroll
    for (int e = 0; e < 8; e++) {
        float p = (vals[e] > -1.0e38f) ? __expf(vals[e] - rmax) : 0.0f;
        vals[e] = p;
        sloc += p;
    }
    red[t] = sloc; __syncthreads();
    for (int s = 128; s > 0; s >>= 1) {
        if (t < s) red[t] += red[t + s];
        __syncthreads();
    }
    float inv = maskq / fmaxf(red[0], 1e-30f);
#pragma unroll
    for (int e = 0; e < 8; e++) rp[t + e * 256] = vals[e] * inv;
}

// K4: out[b] = attn[b] @ text1[b] + bias   (attn already query-masked)
__global__ void __launch_bounds__(256)
k_gemm_out(const float* __restrict__ attn, const u16* __restrict__ t1bfT,
           const float* __restrict__ bias, float* __restrict__ out) {
    __shared__ __align__(16) u16 lA[2][128 * LDSTR];
    __shared__ __align__(16) u16 lB[2][128 * LDSTR];
    int b = blockIdx.z;
    const float* A  = attn  + (size_t)b * SS * SS;
    const u16*   Bt = t1bfT + (size_t)b * DD * SS;   // [D][S], K-contiguous
    float*       C  = out   + (size_t)b * SS * DD;
    int mBase = blockIdx.y * 128;   // over S
    int nBase = blockIdx.x * 128;   // over D
    v8f acc[2][4] = {};

    TileF32 ta = fetch_f32(A, SS, mBase, 0);
    TileB16 tb = fetch_bf16(Bt, SS, nBase, 0);
    store_f32(ta, lA[0]); store_bf16(tb, lB[0]);
    __syncthreads();
    int cur = 0;
    for (int kk = 0; kk < SS; kk += 32) {
        bool more = (kk + 32) < SS;
        TileF32 na;
        TileB16 nb;
        if (more) {
            na = fetch_f32(A, SS, mBase, kk + 32);
            nb = fetch_bf16(Bt, SS, nBase, kk + 32);
        }
        mma_tile(lA[cur], lB[cur], acc);
        if (more) {
            store_f32(na, lA[cur ^ 1]);
            store_bf16(nb, lB[cur ^ 1]);
        }
        __syncthreads();
        cur ^= 1;
    }

    int lane = threadIdx.x & 31;
    int wid  = threadIdx.x >> 5;
    int wm = (wid >> 1) * 32, wn = (wid & 1) * 64;
#pragma unroll
    for (int i = 0; i < 2; i++)
#pragma unroll
        for (int j = 0; j < 4; j++) {
            int col  = nBase + wn + j * 16 + (lane & 15);
            int row0 = mBase + wm + i * 16 + ((lane >> 4) & 1) * 8;
            float bv = bias[col];
#pragma unroll
            for (int e = 0; e < 8; e++)
                C[(size_t)(row0 + e) * DD + col] = acc[i][j][e] + bv;
        }
}

// ---- launch ----------------------------------------------------------------

extern "C" void kernel_launch(void* const* d_in, const int* in_sizes, int n_in,
                              void* d_out, int out_size, void* d_ws, size_t ws_size,
                              hipStream_t stream) {
    const float* text  = (const float*)d_in[0];
    const float* text1 = (const float*)d_in[1];
    const float* mask  = (const float*)d_in[2];
    const float* W     = (const float*)d_in[3];
    const float* bias  = (const float*)d_in[4];

    float* out  = (float*)d_out;
    float* attn = out + (size_t)BB * SS * DD;   // second output region; also logits scratch

    u16* qbf   = (u16*)d_ws;                         // B*S*D bf16
    u16* t1bf  = qbf   + (size_t)BB * SS * DD;       // B*S*D bf16
    u16* t1bfT = t1bf  + (size_t)BB * SS * DD;       // B*D*S bf16

    dim3 blk(256);
    k_prep_text1<<<dim3(DD / 64, SS / 64, BB), blk, 0, stream>>>(text1, t1bf, t1bfT);
    k_gemm_q<<<dim3(DD / 128, (BB * SS) / 128), blk, 0, stream>>>(text, W, qbf);
    k_gemm_logits<<<dim3(SS / 128, SS / 128, BB), blk, 0, stream>>>(qbf, t1bf, attn);
    k_softmax<<<dim3(BB * SS), blk, 0, stream>>>(attn, mask);
    k_gemm_out<<<dim3(DD / 128, SS / 128, BB), blk, 0, stream>>>(attn, t1bfT, bias, out);
}
